// AttentionLayer_50886772523125
// MI455X (gfx1250) — matrix operand order repaired and checked by reading, not compiled, not run on hardware
//
#include <hip/hip_runtime.h>
#include <hip/hip_bf16.h>

typedef __attribute__((ext_vector_type(16))) __bf16       v16bf;
typedef __attribute__((ext_vector_type(8)))  float        v8f;
typedef __attribute__((ext_vector_type(4)))  float        v4f;
typedef __attribute__((ext_vector_type(4)))  unsigned int v4u;
typedef __attribute__((ext_vector_type(8)))  int          i32x8;
typedef __attribute__((ext_vector_type(4)))  int          i32x4;

union FragU { v4u u[2]; v16bf v; };

#define BB 4
#define SS 2048
#define DD 1024
#define FF 1024
#define HH 16
#define DH 64
#define FH 64
#define QT 128          // query rows per block (8 waves x 16 rows)
#define KT 64           // kv rows per tile
#define NW 8            // waves per block
#define LDP 72          // padded LDS row (halves); 144 B (TDM pad keeps 16B align)
#define LOG2E 1.4426950408889634f

__device__ __forceinline__ unsigned int pk_bf16(float a, float b) {
  unsigned int ua = __float_as_uint(a);
  unsigned int ub = __float_as_uint(b);
  ua += 0x7FFFu + ((ua >> 16) & 1u);   // RNE truncate to bf16
  ub += 0x7FFFu + ((ub >> 16) & 1u);
  return (ua >> 16) | (ub & 0xFFFF0000u);
}

__device__ __forceinline__ unsigned short bf16b(float f) {
  unsigned int u = __float_as_uint(f);
  u += 0x7FFFu + ((u >> 16) & 1u);
  return (unsigned short)(u >> 16);
}

__device__ __forceinline__ v8f wmma_bf16(v16bf a, v16bf b, v8f c) {
  return __builtin_amdgcn_wmma_f32_16x16x32_bf16(false, a, false, b,
                                                 (short)0, c, false, false);
}

// ---- Tensor Data Mover: 2D bf16 tile (64 rows x 64 elem) -> padded LDS ----
// D# per CDNA5 ISA 8.3-8.5. Row = 128B; pad_interval=4 (128B), pad_amount=3
// (16B) -> LDS row stride 144B = LDP halves. data_size=1 (2 bytes).
__device__ __forceinline__ void tdm_load_tile(unsigned lds_addr, const void* gptr,
                                              unsigned tensor_d0, unsigned tensor_d1,
                                              unsigned stride0) {
  const unsigned long long ga = (unsigned long long)gptr;
  v4u g0;
  g0[0] = 1u;                                   // count=1, user descriptor
  g0[1] = lds_addr;                             // LDS byte address
  g0[2] = (unsigned)(ga & 0xFFFFFFFFull);       // global_addr[31:0]
  g0[3] = (unsigned)((ga >> 32) & 0x01FFFFFFull) | (2u << 30);  // [56:32] | type=2
  i32x8 g1;
  g1[0] = (int)((1u << 16) | (1u << 20) | (4u << 22) | (3u << 25)); // ds=2B,pad en/int/amt
  g1[1] = (int)((tensor_d0 & 0xFFFFu) << 16);                       // tensor_dim0 lo
  g1[2] = (int)((tensor_d0 >> 16) | ((tensor_d1 & 0xFFFFu) << 16)); // td0 hi | td1 lo
  g1[3] = (int)((tensor_d1 >> 16) | (64u << 16));                   // td1 hi | tile_dim0=64
  g1[4] = (int)64u;                                                 // tile_dim1=64, tile_dim2=0
  g1[5] = (int)stride0;                                             // tensor_dim0_stride lo
  g1[6] = 0;
  g1[7] = 0;
  i32x4 z4 = {0, 0, 0, 0};
#if __clang_major__ >= 23
  i32x8 z8 = {0, 0, 0, 0, 0, 0, 0, 0};
  __builtin_amdgcn_tensor_load_to_lds(g0, g1, z4, z4, z8, 0);
#else
  __builtin_amdgcn_tensor_load_to_lds(g0, g1, z4, z4, 0);
#endif
}

// =====================  prep: diag-scale + f32->bf16  =====================
// Qb/Kb: [b][h][s][64] row-major (Q pre-scaled by wq*0.125*log2e).
// Vtb:   [b][h][f][s]  (transposed so AV B-fragments are contiguous).
__global__ __launch_bounds__(256)
void prep_kernel(const float* __restrict__ XQ, const float* __restrict__ XK,
                 const float* __restrict__ XV, const float* __restrict__ WQ,
                 const float* __restrict__ WK, const float* __restrict__ WV,
                 unsigned short* __restrict__ Qb, unsigned short* __restrict__ Kb,
                 unsigned short* __restrict__ Vtb) {
  __shared__ float wqs[DH], wks[DH], wvs[FH];
  const int tid = threadIdx.x;
  const int nSB = SS / QT;
  const int blk = blockIdx.x;
  const int sc  = blk % nSB;
  const int h   = (blk / nSB) % HH;
  const int b   = blk / (nSB * HH);
  const int s0  = sc * QT;

  if (tid < DH) {
    wqs[tid] = WQ[h * DH * DH + tid * DH + tid] * (0.125f * LOG2E);
    wks[tid] = WK[h * DH * DH + tid * DH + tid];
    wvs[tid] = WV[h * FH * FH + tid * FH + tid];
  }
  __syncthreads();

  {  // Q and K rows: 2 threads/row, 32 cols each
    const int r  = tid >> 1;
    const int c0 = (tid & 1) * 32;
    const float* srcq = XQ + ((size_t)b * SS + s0 + r) * DD + h * DH + c0;
    const float* srck = XK + ((size_t)b * SS + s0 + r) * DD + h * DH + c0;
    unsigned short* dq = Qb + ((size_t)(b * HH + h) * SS + s0 + r) * 64 + c0;
    unsigned short* dk = Kb + ((size_t)(b * HH + h) * SS + s0 + r) * 64 + c0;
#pragma unroll
    for (int i = 0; i < 32; i += 8) {
      v4f f0 = *(const v4f*)(srcq + i);
      v4f f1 = *(const v4f*)(srcq + i + 4);
      v4u pq = { pk_bf16(f0.x * wqs[c0 + i + 0], f0.y * wqs[c0 + i + 1]),
                 pk_bf16(f0.z * wqs[c0 + i + 2], f0.w * wqs[c0 + i + 3]),
                 pk_bf16(f1.x * wqs[c0 + i + 4], f1.y * wqs[c0 + i + 5]),
                 pk_bf16(f1.z * wqs[c0 + i + 6], f1.w * wqs[c0 + i + 7]) };
      *(v4u*)(dq + i) = pq;
      f0 = *(const v4f*)(srck + i);
      f1 = *(const v4f*)(srck + i + 4);
      v4u pk2 = { pk_bf16(f0.x * wks[c0 + i + 0], f0.y * wks[c0 + i + 1]),
                  pk_bf16(f0.z * wks[c0 + i + 2], f0.w * wks[c0 + i + 3]),
                  pk_bf16(f1.x * wks[c0 + i + 4], f1.y * wks[c0 + i + 5]),
                  pk_bf16(f1.z * wks[c0 + i + 6], f1.w * wks[c0 + i + 7]) };
      *(v4u*)(dk + i) = pk2;
    }
  }
  {  // V transposed: thread owns one f row segment, 32 s values (coalesced writes)
    const int f   = tid >> 2;
    const int sof = (tid & 3) * 32;
    const float wv = wvs[f];
    const float* src = XV + ((size_t)b * SS + s0 + sof) * FF + h * FH + f;
    unsigned short* dst = Vtb + ((size_t)(b * HH + h) * 64 + f) * SS + s0 + sof;
    unsigned int pd[16];
#pragma unroll
    for (int i = 0; i < 16; ++i) {
      const float a0 = src[(size_t)(2 * i) * FF] * wv;
      const float a1 = src[(size_t)(2 * i + 1) * FF] * wv;
      pd[i] = pk_bf16(a0, a1);
    }
#pragma unroll
    for (int c = 0; c < 4; ++c) {
      v4u p = { pd[c * 4 + 0], pd[c * 4 + 1], pd[c * 4 + 2], pd[c * 4 + 3] };
      *(v4u*)(dst + c * 8) = p;
    }
  }
}

// ====================  main: TDM double-buffered flash attention  ====================
__global__ __launch_bounds__(256)
void fattn_main(const unsigned short* __restrict__ Qb,
                const unsigned short* __restrict__ Kb,
                const unsigned short* __restrict__ Vtb,
                const float* __restrict__ Wo, float* __restrict__ Y) {
  __shared__ __align__(16) unsigned short Ks[2][KT * LDP];    // K tiles (TDM)
  __shared__ __align__(16) unsigned short Vts[2][FH * LDP];   // V^T tiles (TDM)
  __shared__ __align__(16) unsigned short Ps[NW * 16 * LDP];  // per-wave P scratch
  __shared__ float ods[FH];

  const int tid  = threadIdx.x;
  const int lane = tid & 31;
  const int w    = tid >> 5;
  const int nQB  = SS / QT;
  const int blk  = blockIdx.x;
  const int qb   = blk % nQB;
  const int h    = (blk / nQB) % HH;
  const int b    = blk / (nQB * HH);
  const int q0   = qb * QT;

  if (tid < FH) {
    const int f = h * FH + tid;
    ods[tid] = Wo[(size_t)f * FF + f];
  }

  // ---- per-wave Q A-fragments straight from global bf16 ----
  v16bf aQ0, aQ1;
  {
    const int m = lane & 15, sel = lane >> 4;
    const unsigned short* p =
        Qb + ((size_t)(b * HH + h) * SS + q0 + w * 16 + m) * 64 + sel * 8;
    FragU f;
    f.u[0] = *(const v4u*)(p);       f.u[1] = *(const v4u*)(p + 16); aQ0 = f.v;
    f.u[0] = *(const v4u*)(p + 32);  f.u[1] = *(const v4u*)(p + 48); aQ1 = f.v;
  }

  v8f accO[4] = {{}, {}, {}, {}};
  float mrow[8], lrow[8];
#pragma unroll
  for (int j = 0; j < 8; ++j) { mrow[j] = -1e30f; lrow[j] = 0.f; }

  const unsigned short* kbase = Kb + (size_t)(b * HH + h) * SS * 64;
  const unsigned short* vbase = Vtb + (size_t)(b * HH + h) * 64 * SS;
  const unsigned ldsK[2] = { (unsigned)(unsigned long long)(const void*)&Ks[0][0],
                             (unsigned)(unsigned long long)(const void*)&Ks[1][0] };
  const unsigned ldsV[2] = { (unsigned)(unsigned long long)(const void*)&Vts[0][0],
                             (unsigned)(unsigned long long)(const void*)&Vts[1][0] };

  // prologue: TDM-issue tile 0 (wave0 -> K, wave1 -> V; EXEC ignored, 1 issue/wave)
  if (w == 0) tdm_load_tile(ldsK[0], kbase, 64u, (unsigned)SS, 64u);
  if (w == 1) tdm_load_tile(ldsV[0], vbase, (unsigned)SS, 64u, (unsigned)SS);

  const int NT = SS / KT;
  for (int i = 0; i < NT; ++i) {
    const int cur = i & 1;
    if (i + 1 < NT) {                       // prefetch next tile into other buffer
      const int nxt = (i + 1) & 1;
      const int kvn = (i + 1) * KT;
      if (w == 0) tdm_load_tile(ldsK[nxt], kbase + (size_t)kvn * 64, 64u, (unsigned)SS, 64u);
      if (w == 1) tdm_load_tile(ldsV[nxt], vbase + kvn, (unsigned)SS, 64u, (unsigned)SS);
      __builtin_amdgcn_s_wait_tensorcnt(1); // current tile complete (in-order)
    } else {
      __builtin_amdgcn_s_wait_tensorcnt(0);
    }
    __syncthreads();                        // publish current tile to all waves

    const unsigned short* ksb = &Ks[cur][0];
    const unsigned short* vsb = &Vts[cur][0];

    // ---- S = Q * K^T : 4 N-tiles x 2 K-slabs = 8 WMMA ----
    v8f accS[4] = {{}, {}, {}, {}};
#pragma unroll
    for (int t = 0; t < 4; ++t) {
      const int n = lane & 15, sel = lane >> 4;
      const unsigned short* p = ksb + (t * 16 + n) * LDP + sel * 8;
      FragU f;
      f.u[0] = *(const v4u*)(p);      f.u[1] = *(const v4u*)(p + 16);
      const v16bf bk0 = f.v;
      f.u[0] = *(const v4u*)(p + 32); f.u[1] = *(const v4u*)(p + 48);
      const v16bf bk1 = f.v;
      accS[t] = wmma_bf16(aQ0, bk0, accS[t]);
      accS[t] = wmma_bf16(aQ1, bk1, accS[t]);
    }

    // ---- online softmax (scores in log2 domain; wave32 16-lane reductions) ----
    float pr[4][8];
#pragma unroll
    for (int j = 0; j < 8; ++j) {
      float v = fmaxf(fmaxf(accS[0][j], accS[1][j]), fmaxf(accS[2][j], accS[3][j]));
      v = fmaxf(v, __shfl_xor(v, 1, 32));
      v = fmaxf(v, __shfl_xor(v, 2, 32));
      v = fmaxf(v, __shfl_xor(v, 4, 32));
      v = fmaxf(v, __shfl_xor(v, 8, 32));
      const float mn    = fmaxf(mrow[j], v);
      const float alpha = __builtin_amdgcn_exp2f(mrow[j] - mn);
      mrow[j] = mn;
      float rs = 0.f;
#pragma unroll
      for (int t = 0; t < 4; ++t) {
        const float p = __builtin_amdgcn_exp2f(accS[t][j] - mn);
        pr[t][j] = p;
        rs += p;
      }
      rs += __shfl_xor(rs, 1, 32);
      rs += __shfl_xor(rs, 2, 32);
      rs += __shfl_xor(rs, 4, 32);
      rs += __shfl_xor(rs, 8, 32);
      lrow[j] = lrow[j] * alpha + rs;
#pragma unroll
      for (int t = 0; t < 4; ++t) accO[t][j] *= alpha;
    }

    // ---- P: D-layout -> row-major LDS -> A-layout ----
    unsigned short* pw = Ps + w * 16 * LDP;
    {
      const int n = lane & 15, rb = (lane >> 4) * 8;
#pragma unroll
      for (int t = 0; t < 4; ++t)
#pragma unroll
        for (int j = 0; j < 8; ++j)
          pw[(rb + j) * LDP + t * 16 + n] = bf16b(pr[t][j]);
    }
    v16bf aP0, aP1;
    {
      const int m = lane & 15, sel = lane >> 4;
      const unsigned short* p = pw + m * LDP + sel * 8;
      FragU f;
      f.u[0] = *(const v4u*)(p);      f.u[1] = *(const v4u*)(p + 16); aP0 = f.v;
      f.u[0] = *(const v4u*)(p + 32); f.u[1] = *(const v4u*)(p + 48); aP1 = f.v;
    }

    // ---- O += P * V : 4 N-tiles x 2 K-slabs = 8 WMMA ----
#pragma unroll
    for (int t = 0; t < 4; ++t) {
      const int n = lane & 15, sel = lane >> 4;
      const unsigned short* p = vsb + (t * 16 + n) * LDP + sel * 8;
      FragU f;
      f.u[0] = *(const v4u*)(p);      f.u[1] = *(const v4u*)(p + 16);
      const v16bf bv0 = f.v;
      f.u[0] = *(const v4u*)(p + 32); f.u[1] = *(const v4u*)(p + 48);
      const v16bf bv1 = f.v;
      accO[t] = wmma_bf16(aP0, bv0, accO[t]);
      accO[t] = wmma_bf16(aP1, bv1, accO[t]);
    }
    __syncthreads();   // compute done: buffer may be overwritten by next issue
  }

  // ---- epilogue: normalize, apply o_diag, store f32 ----
  {
    const int n = lane & 15, rb = (lane >> 4) * 8;
#pragma unroll
    for (int j = 0; j < 8; ++j) {
      const float invl = 1.f / lrow[j];
      const int row = q0 + w * 16 + rb + j;
      float* dst = Y + ((size_t)b * SS + row) * FF + h * FH;
#pragma unroll
      for (int t = 0; t < 4; ++t)
        dst[t * 16 + n] = accO[t][j] * invl * ods[t * 16 + n];
    }
  }
}

extern "C" void kernel_launch(void* const* d_in, const int* in_sizes, int n_in,
                              void* d_out, int out_size, void* d_ws, size_t ws_size,
                              hipStream_t stream) {
  (void)in_sizes; (void)n_in; (void)out_size; (void)ws_size;
  const float* XQ = (const float*)d_in[0];
  const float* XK = (const float*)d_in[1];
  const float* XV = (const float*)d_in[2];
  const float* WQ = (const float*)d_in[3];
  const float* WK = (const float*)d_in[4];
  const float* WV = (const float*)d_in[5];
  const float* Wo = (const float*)d_in[6];
  float* Y = (float*)d_out;

  unsigned short* Qb  = (unsigned short*)d_ws;
  unsigned short* Kb  = Qb + (size_t)BB * HH * SS * 64;
  unsigned short* Vtb = Kb + (size_t)BB * HH * SS * 64;

  dim3 block(256);
  dim3 grid(BB * HH * (SS / QT));   // 1024 blocks
  prep_kernel<<<grid, block, 0, stream>>>(XQ, XK, XV, WQ, WK, WV, Qb, Kb, Vtb);
  fattn_main<<<grid, block, 0, stream>>>(Qb, Kb, Vtb, Wo, Y);
}
